// IMem_Pool_31078383354674
// MI455X (gfx1250) — matrix operand order repaired and checked by reading, not compiled, not run on hardware
//
#include <hip/hip_runtime.h>
#include <hip/hip_bf16.h>

typedef _Float16 v16h __attribute__((ext_vector_type(16)));
typedef float    v8f  __attribute__((ext_vector_type(8)));
typedef _Float16 h8   __attribute__((ext_vector_type(8)));

#define BATCH   128
#define NNODE   512
#define NN      65536      // BATCH*NNODE
#define FDIM    128
#define CDIM    384        // 3*HID
#define NEDGE   1048576    // BATCH*8192
#define NCENT   15         // HEADS*KC = 5*3

// ---------------------------------------------------------------- f32 -> f16
__global__ void k_cvt(const float* __restrict__ in, _Float16* __restrict__ out, int n) {
    int i = blockIdx.x * blockDim.x + threadIdx.x;
    if (i < n) out[i] = (_Float16)in[i];
}

// ---------------------------------------------------------------- W (128x128 f32, KxN)
// -> f16 pre-swizzled into WMMA B-fragment order:
// out[ ((nt*4 + kc)*32 + lane)*16 + j ] = W[ (kc*32 + (j>>3)*16 + (lane>>4)*8 + (j&7))*128 + nt*16 + (lane&15) ]
// so each lane's 16-half B fragment is one contiguous 32B block.
__global__ void k_cvt_wswz(const float* __restrict__ W, _Float16* __restrict__ out) {
    const int i    = blockIdx.x * blockDim.x + threadIdx.x;   // 0..16383
    const int j    = i & 15;
    const int lane = (i >> 4) & 31;
    const int kc   = (i >> 9) & 3;
    const int nt   = i >> 11;
    const int K = kc * 32 + (j >> 3) * 16 + (lane >> 4) * 8 + (j & 7);
    const int n = nt * 16 + (lane & 15);
    out[i] = (_Float16)W[K * 128 + n];
}

// -------------------------------------------------- WMMA GEMM: C = A(f16) @ B(f16 swz)
// A: Nn x 128 row-major, Bsw: fragment-ordered (see k_cvt_wswz), C: Nn x 128 f32.
// Block = 256 thr (8 waves); each wave computes a 16-row x 128-col strip.
__global__ void k_gemm128(const _Float16* __restrict__ A,
                          const _Float16* __restrict__ Bsw,
                          float* __restrict__ C) {
    const int wave = threadIdx.x >> 5;
    const int lane = threadIdx.x & 31;
    const int mrow = lane & 15;
    const int hi   = lane >> 4;                 // 0: lanes 0-15, 1: lanes 16-31
    const size_t row0 = (size_t)blockIdx.x * 128 + (size_t)wave * 16;
    const _Float16* Arow = A + (row0 + mrow) * 128;

    // Preload 4 A fragments (K chunks of 32). Per ISA layout:
    // halves 0..7 -> K = k0 + hi*8 + j ; halves 8..15 -> K = k0 + 16 + hi*8 + j.
    v16h af[4];
#pragma unroll
    for (int kc = 0; kc < 4; ++kc) {
        const int k0 = kc * 32;
        h8 lo = *(const h8*)(Arow + k0 + hi * 8);
        h8 up = *(const h8*)(Arow + k0 + 16 + hi * 8);
        af[kc] = __builtin_shufflevector(lo, up, 0, 1, 2, 3, 4, 5, 6, 7,
                                         8, 9, 10, 11, 12, 13, 14, 15);
    }

#pragma unroll
    for (int nt = 0; nt < 8; ++nt) {
        v8f acc = {};
#pragma unroll
        for (int kc = 0; kc < 4; ++kc) {
            const v16h bf = *(const v16h*)(Bsw + ((size_t)(nt * 4 + kc) * 32 + lane) * 16);
            acc = __builtin_amdgcn_wmma_f32_16x16x32_f16(
                false, af[kc], false, bf, (short)0, acc, false, false);
        }
        float* Cp = C + (row0 + (size_t)hi * 8) * 128 + nt * 16 + mrow;
#pragma unroll
        for (int v = 0; v < 8; ++v) Cp[(size_t)v * 128] = acc[v];   // M = hi*8 + v
    }
}

// -------------------------------------------------- edge scatter: agg[dst] += m[src]
// one wave per edge: 32 lanes x float4 = 128 features
__global__ void k_scatter(const int* __restrict__ src, const int* __restrict__ dst,
                          const float* __restrict__ m, float* __restrict__ agg, int E) {
    const int lane = threadIdx.x & 31;
    const int gw   = (blockIdx.x * blockDim.x + threadIdx.x) >> 5;
    const int nw   = (gridDim.x * blockDim.x) >> 5;
    for (int e = gw; e < E; e += nw) {
        const int s = src[e], d = dst[e];
        const float4 v = *(const float4*)(m + (size_t)s * 128 + lane * 4);
        float* p = agg + (size_t)d * 128 + lane * 4;
        atomicAdd(p + 0, v.x); atomicAdd(p + 1, v.y);
        atomicAdd(p + 2, v.z); atomicAdd(p + 3, v.w);
    }
}

// -------------------------------------------------- h = relu(agg + b); optional f16 copy
__global__ void k_bias_relu(const float* __restrict__ agg, const float* __restrict__ bias,
                            float* __restrict__ h, _Float16* __restrict__ hh, int total) {
    int i = blockIdx.x * blockDim.x + threadIdx.x;
    if (i >= total) return;
    float v = fmaxf(agg[i] + bias[i & 127], 0.f);
    h[i] = v;
    if (hh) hh[i] = (_Float16)v;
}

// -------------------------------------------------- BN column sums (384 cols over X=concat)
__global__ void k_bn_stats(const float* __restrict__ h1, const float* __restrict__ h2,
                           const float* __restrict__ h3,
                           float* __restrict__ sums, float* __restrict__ sumsq) {
    const int c = threadIdx.x;                       // 0..383
    const int rows = NN / gridDim.x;
    const int r0 = blockIdx.x * rows;
    const float* base = (c < 128) ? h1 : (c < 256 ? h2 : h3);
    const int cc = c & 127;
    float s = 0.f, q = 0.f;
    for (int r = r0; r < r0 + rows; ++r) {
        float v = base[(size_t)r * 128 + cc];
        s += v; q += v * v;
    }
    atomicAdd(&sums[c], s);
    atomicAdd(&sumsq[c], q);
}

__global__ void k_bn_final(const float* __restrict__ sums, const float* __restrict__ sumsq,
                           const float* __restrict__ gamma, const float* __restrict__ beta,
                           float* __restrict__ scale, float* __restrict__ shift) {
    int c = threadIdx.x;
    float mu  = sums[c] / (float)NN;
    float var = sumsq[c] / (float)NN - mu * mu;
    float sc  = gamma[c] * rsqrtf(var + 1e-5f);
    scale[c] = sc;
    shift[c] = beta[c] - mu * sc;
}

// -------------------------------------------------- per-node soft assignment (wave/node)
__global__ void k_dist(const float* __restrict__ h1, const float* __restrict__ h2,
                       const float* __restrict__ h3, const float* __restrict__ scale,
                       const float* __restrict__ shift, const float* __restrict__ k2,
                       float* __restrict__ Sf) {
    const int gw = (blockIdx.x * blockDim.x + threadIdx.x) >> 5;   // node
    const int lane = threadIdx.x & 31;
    if (gw >= NN) return;
    float acc[NCENT];
#pragma unroll
    for (int t = 0; t < NCENT; ++t) acc[t] = 0.f;
#pragma unroll
    for (int j = 0; j < 12; ++j) {
        const int c = lane + j * 32;
        const float* base = (c < 128) ? h1 : (c < 256 ? h2 : h3);
        const float xv = base[(size_t)gw * 128 + (c & 127)] * scale[c] + shift[c];
#pragma unroll
        for (int t = 0; t < NCENT; ++t) {
            float d = k2[t * CDIM + c] - xv;
            acc[t] += d * d;
        }
    }
#pragma unroll
    for (int off = 16; off > 0; off >>= 1) {
#pragma unroll
        for (int t = 0; t < NCENT; ++t) acc[t] += __shfl_down(acc[t], off);
    }
    if (lane == 0) {
        float S0 = 0.f, S1 = 0.f, S2 = 0.f;
#pragma unroll
        for (int h = 0; h < 5; ++h) {
            float d0 = 1.f / (1.f + acc[h * 3 + 0]);   // TAU=1 -> (1+d2)^-1
            float d1 = 1.f / (1.f + acc[h * 3 + 1]);
            float d2 = 1.f / (1.f + acc[h * 3 + 2]);
            float inv = 1.f / (d0 + d1 + d2);
            S0 += d0 * inv; S1 += d1 * inv; S2 += d2 * inv;
        }
        float* o = Sf + (size_t)gw * 3;
        o[0] = S0; o[1] = S1; o[2] = S2;
    }
}

// -------------------------------------------------- Sagg[src] += Sf[dst], degree counts
__global__ void k_sagg(const int* __restrict__ src, const int* __restrict__ dst,
                       const float* __restrict__ Sf, float* __restrict__ Sagg,
                       float* __restrict__ deg, int E) {
    int e = blockIdx.x * blockDim.x + threadIdx.x;
    if (e >= E) return;
    const int s = src[e], d = dst[e];
    const float* f = Sf + (size_t)d * 3;
    float* o = Sagg + (size_t)s * 3;
    atomicAdd(o + 0, f[0]); atomicAdd(o + 1, f[1]); atomicAdd(o + 2, f[2]);
    atomicAdd(deg + s, 1.f); atomicAdd(deg + d, 1.f);
}

// -------------------------------------------------- softmax/argmax + loss partials
// stats: [0]=counts0 [1]=counts1 [2]=A0(sum mf*ce) [3]=A1 [4]=sumloss
__global__ void k_assign(const float* __restrict__ Sagg, const float* __restrict__ deg,
                         const int* __restrict__ y, float* __restrict__ Sfin,
                         float* __restrict__ stats) {
    __shared__ float ls[5];
    if (threadIdx.x < 5) ls[threadIdx.x] = 0.f;
    __syncthreads();
    const int i = blockIdx.x * blockDim.x + threadIdx.x;
    if (i < NN) {
        float dg = deg[i] * 0.5f;
        if (dg == 0.f) dg = 1.f;
        const float inv = 1.f / dg;
        const float r0 = Sagg[(size_t)i * 3 + 0] * inv;
        const float r1 = Sagg[(size_t)i * 3 + 1] * inv;
        const float r2 = Sagg[(size_t)i * 3 + 2] * inv;
        const float m = fmaxf(r0, fmaxf(r1, r2));
        const float e0 = expf(r0 - m), e1 = expf(r1 - m), e2 = expf(r2 - m);
        const float is = 1.f / (e0 + e1 + e2);
        const float g0 = e0 * is, g1 = e1 * is, g2 = e2 * is;
        int a = 0; float ga = g0;
        if (g1 > ga) { a = 1; ga = g1; }
        if (g2 > ga) { a = 2; ga = g2; }
        const float s0 = (a == 0) ? g0 : 0.f;
        const float s1 = (a == 1) ? g1 : 0.f;
        const float s2 = (a == 2) ? g2 : 0.f;
        float* o = Sfin + (size_t)i * 3;
        o[0] = s0; o[1] = s1; o[2] = s2;
        const int yb = y[i >> 9];
        const float mm = fmaxf(s0, s1);
        const float lse = mm + logf(expf(s0 - mm) + expf(s1 - mm));
        const float ce = lse - (yb ? s1 : s0);
        atomicAdd(&ls[4], s0 + s1);
        if (a < 2) { atomicAdd(&ls[yb], 1.f); atomicAdd(&ls[2 + yb], ce); }
    }
    __syncthreads();
    if (threadIdx.x < 5) atomicAdd(&stats[threadIdx.x], ls[threadIdx.x]);
}

// -------------------------------------------------- pooled xp[b,k,c] = sum_n Sfin*Xn
__global__ void k_pool(const float* __restrict__ h1, const float* __restrict__ h2,
                       const float* __restrict__ h3, const float* __restrict__ scale,
                       const float* __restrict__ shift, const float* __restrict__ Sfin,
                       float* __restrict__ xp) {
    const int b = blockIdx.x;
    const int c = threadIdx.x;                 // 0..383
    const float* base = (c < 128) ? h1 : (c < 256 ? h2 : h3);
    const int cc = c & 127;
    const float sc = scale[c], sh = shift[c];
    float a0 = 0.f, a1 = 0.f, a2 = 0.f;
    const size_t node0 = (size_t)b * NNODE;
    for (int n = 0; n < NNODE; ++n) {
        const size_t node = node0 + n;
        const float xv = base[node * 128 + cc] * sc + sh;
        const float* s = Sfin + node * 3;
        a0 += s[0] * xv; a1 += s[1] * xv; a2 += s[2] * xv;
    }
    float* o = xp + (size_t)b * (3 * CDIM);
    o[c] = a0; o[CDIM + c] = a1; o[2 * CDIM + c] = a2;
}

// -------------------------------------------------- xf[b, k*16+o] = xp[b,k,:] @ lin_pool_W
__global__ void k_pool_lin(const float* __restrict__ xp, const float* __restrict__ W,
                           float* __restrict__ xf) {
    const int b = blockIdx.x;
    const int t = threadIdx.x;                 // 0..31 : k = t>>4, o = t&15
    const float* r = xp + (size_t)b * (3 * CDIM) + (size_t)(t >> 4) * CDIM;
    const int o = t & 15;
    float s = 0.f;
    for (int c = 0; c < CDIM; ++c) s += r[c] * W[c * 16 + o];
    xf[b * 32 + t] = s;
}

// -------------------------------------------------- head MLP + log_softmax
__global__ void k_mlp(const float* __restrict__ xf, const float* __restrict__ W1,
                      const float* __restrict__ b1, const float* __restrict__ W2,
                      const float* __restrict__ b2, float* __restrict__ out) {
    __shared__ float row[32];
    __shared__ float hid[128];
    __shared__ float lg[2];
    const int b = blockIdx.x, t = threadIdx.x;
    if (t < 32) row[t] = xf[b * 32 + t];
    __syncthreads();
    float a = b1[t];
    for (int i = 0; i < 32; ++i) a += row[i] * W1[i * 128 + t];
    hid[t] = fmaxf(a, 0.f);
    __syncthreads();
    if (t < 2) {
        float s = b2[t];
        for (int i = 0; i < 128; ++i) s += hid[i] * W2[i * 2 + t];
        lg[t] = s;
    }
    __syncthreads();
    if (t < 2) {
        float m = fmaxf(lg[0], lg[1]);
        float lse = m + logf(expf(lg[0] - m) + expf(lg[1] - m));
        out[b * 2 + t] = lg[t] - lse;
    }
}

// -------------------------------------------------- final scalar loss
__global__ void k_final(const float* __restrict__ stats, float* __restrict__ out_loss) {
    const float c0 = stats[0], c1 = stats[1];
    const float bc0 = 1.f + c0, bc1 = 1.f + c1;
    const float mx = fmaxf(bc0, bc1);
    const float w0 = mx / (bc0 + 0.001f), w1 = mx / (bc1 + 0.001f);
    const float num = w0 * stats[2] + w1 * stats[3];
    const float den = w0 * c0 + w1 * c1;
    const float cel = num / fmaxf(den, 1e-12f);
    *out_loss = 10.f * cel + 0.01f * (stats[4] / (float)BATCH);
}

// ================================================================ launch
extern "C" void kernel_launch(void* const* d_in, const int* in_sizes, int n_in,
                              void* d_out, int out_size, void* d_ws, size_t ws_size,
                              hipStream_t stream) {
    (void)in_sizes; (void)n_in; (void)out_size; (void)ws_size;
    const float* x     = (const float*)d_in[0];
    const int*   ei    = (const int*)d_in[1];
    const int*   src   = ei;
    const int*   dst   = ei + NEDGE;
    const int*   y     = (const int*)d_in[2];
    const float* W1    = (const float*)d_in[3];
    const float* b1    = (const float*)d_in[4];
    const float* W2    = (const float*)d_in[5];
    const float* b2    = (const float*)d_in[6];
    const float* W3    = (const float*)d_in[7];
    const float* b3    = (const float*)d_in[8];
    const float* gamma = (const float*)d_in[9];
    const float* beta  = (const float*)d_in[10];
    const float* k2    = (const float*)d_in[11];   // (5,3,384) == (15,384)
    const float* poolW = (const float*)d_in[12];
    const float* l1W   = (const float*)d_in[13];
    const float* l1b   = (const float*)d_in[14];
    const float* l2W   = (const float*)d_in[15];
    const float* l2b   = (const float*)d_in[16];
    float* out = (float*)d_out;

    // workspace layout (256B aligned)
    size_t off = 0;
    auto take = [&](size_t bytes) { size_t r = off; off = (off + bytes + 255) & ~(size_t)255; return r; };
    char* ws = (char*)d_ws;
    const size_t FEATB = (size_t)NN * 128 * sizeof(float);           // 32 MB
    float*    mbuf  = (float*)(ws + take(FEATB));
    float*    agg   = (float*)(ws + take(FEATB));
    float*    h1    = (float*)(ws + take(FEATB));
    float*    h2    = (float*)(ws + take(FEATB));
    float*    h3    = (float*)(ws + take(FEATB));
    _Float16* hh    = (_Float16*)(ws + take((size_t)NN * 128 * 2));  // 16 MB f16 GEMM input
    _Float16* Wh    = (_Float16*)(ws + take(3 * 16384 * 2));         // fragment-swizzled
    float*    sums  = (float*)(ws + take(CDIM * sizeof(float)));
    float*    sumsq = (float*)(ws + take(CDIM * sizeof(float)));
    float*    scale = (float*)(ws + take(CDIM * sizeof(float)));
    float*    shift = (float*)(ws + take(CDIM * sizeof(float)));
    float*    Sf    = (float*)(ws + take((size_t)NN * 3 * sizeof(float)));
    float*    Sagg  = (float*)(ws + take((size_t)NN * 3 * sizeof(float)));
    float*    Sfin  = (float*)(ws + take((size_t)NN * 3 * sizeof(float)));
    float*    deg   = (float*)(ws + take((size_t)NN * sizeof(float)));
    float*    xp    = (float*)(ws + take((size_t)BATCH * 3 * CDIM * sizeof(float)));
    float*    xf    = (float*)(ws + take((size_t)BATCH * 32 * sizeof(float)));
    float*    stats = (float*)(ws + take(64));

    const int elemTotal = NN * 128;

    // f16 conversions: x (row-major) and the three weight matrices (fragment-swizzled)
    k_cvt<<<elemTotal / 256, 256, 0, stream>>>(x, hh, elemTotal);
    k_cvt_wswz<<<64, 256, 0, stream>>>(W1, Wh + 0 * 16384);
    k_cvt_wswz<<<64, 256, 0, stream>>>(W2, Wh + 1 * 16384);
    k_cvt_wswz<<<64, 256, 0, stream>>>(W3, Wh + 2 * 16384);

    const float* biases[3] = { b1, b2, b3 };
    float* hs[3] = { h1, h2, h3 };
    for (int l = 0; l < 3; ++l) {
        k_gemm128<<<NN / 128, 256, 0, stream>>>(hh, Wh + (size_t)l * 16384, mbuf);
        hipMemsetAsync(agg, 0, FEATB, stream);
        k_scatter<<<4096, 256, 0, stream>>>(src, dst, mbuf, agg, NEDGE);
        k_bias_relu<<<elemTotal / 256, 256, 0, stream>>>(agg, biases[l], hs[l],
                                                         (l < 2) ? hh : (_Float16*)nullptr,
                                                         elemTotal);
    }

    // BatchNorm stats -> scale/shift
    hipMemsetAsync(sums, 0, CDIM * sizeof(float), stream);
    hipMemsetAsync(sumsq, 0, CDIM * sizeof(float), stream);
    k_bn_stats<<<256, CDIM, 0, stream>>>(h1, h2, h3, sums, sumsq);
    k_bn_final<<<1, CDIM, 0, stream>>>(sums, sumsq, gamma, beta, scale, shift);

    // per-node soft assignment
    k_dist<<<NN / 8, 256, 0, stream>>>(h1, h2, h3, scale, shift, k2, Sf);

    // edge aggregation of assignments + degrees + loss partials
    hipMemsetAsync(Sagg, 0, (size_t)NN * 3 * sizeof(float), stream);
    hipMemsetAsync(deg, 0, (size_t)NN * sizeof(float), stream);
    hipMemsetAsync(stats, 0, 64, stream);
    k_sagg<<<NEDGE / 256, 256, 0, stream>>>(src, dst, Sf, Sagg, deg, NEDGE);
    k_assign<<<NN / 256, 256, 0, stream>>>(Sagg, deg, y, Sfin, stats);

    // pooling + head
    k_pool<<<BATCH, CDIM, 0, stream>>>(h1, h2, h3, scale, shift, Sfin, xp);
    k_pool_lin<<<BATCH, 32, 0, stream>>>(xp, poolW, xf);
    k_mlp<<<BATCH, 128, 0, stream>>>(xf, l1W, l1b, l2W, l2b, out);
    k_final<<<1, 1, 0, stream>>>(stats, out + BATCH * 2);
}
